// Model_17789754540645
// MI455X (gfx1250) — matrix operand order repaired and checked by reading, not compiled, not run on hardware
//
#include <hip/hip_runtime.h>
#include <hip/hip_bf16.h>

// ---------------------------------------------------------------------------
// Fused attention layer for MI455X (gfx1250), wave32 + WMMA f16 path.
// v3: async global->LDS copies via inline asm (global_load_async_to_lds_b128,
//     ASYNCcnt-fenced), double-buffered LDS tiles, preloaded B fragments.
// ---------------------------------------------------------------------------

#define BB 2
#define SS 2048
#define EE 2048
#define HH 16
#define HDD 128

typedef __attribute__((ext_vector_type(16))) _Float16 v16h;
typedef __attribute__((ext_vector_type(8)))  _Float16 v8h;
typedef __attribute__((ext_vector_type(8)))  float    v8f;
typedef __attribute__((ext_vector_type(4)))  float    v4f;

// ---- WMMA wrapper: D = A(16x32 f16) * B(32x16 f16) + C(16x16 f32) ----------
__device__ __forceinline__ v8f wmma_f16(v16h a, v16h b, v8f c) {
  return __builtin_amdgcn_wmma_f32_16x16x32_f16(
      /*neg_a=*/false, a, /*neg_b=*/false, b,
      /*c_mod=*/(short)0, c, /*reuse_a=*/false, /*reuse_b=*/false);
}

// ---- async global->LDS copy of 16 bytes (ASYNCcnt-tracked) -----------------
// VGLOBAL async encoding: VDST = 32-bit LDS byte address, VADDR = 64-bit
// global address, SADDR = off.  (CDNA5 ISA)
__device__ __forceinline__ void async_copy16(const _Float16* g, _Float16* l) {
  unsigned lds = (unsigned)(size_t)(__attribute__((address_space(3))) _Float16*)l;
  asm volatile("global_load_async_to_lds_b128 %0, %1, off"
               :
               : "v"(lds), "v"(g)
               : "memory");
}

__device__ __forceinline__ void wait_async0() {
  asm volatile("s_wait_asynccnt 0" ::: "memory");
}

// ---- A fragment (16x32, MxK), row-major source with row stride `stride` ----
// Lane holds row M = lane&15.  Lanes 0-15: K=0..7 & 16..23,
// lanes 16-31: K=8..15 & 24..31  -> two contiguous 8-half (16B) loads.
__device__ __forceinline__ v16h load_a_frag(const _Float16* __restrict__ base,
                                            int stride, int lane) {
  const _Float16* p = base + (size_t)(lane & 15) * stride + ((lane >> 4) << 3);
  v8h lo = *(const v8h*)(p);
  v8h hi = *(const v8h*)(p + 16);
  v16h a;
#pragma unroll
  for (int i = 0; i < 8; ++i) { a[i] = lo[i]; a[i + 8] = hi[i]; }
  return a;
}

// ---- B fragment (32x16, KxN) from an N-major buffer (row n, stride halfs) --
// Lane holds column N = lane&15.  Lanes 0-15: K=0..7 & 16..23,
// lanes 16-31: K=8..15 & 24..31  -> two contiguous 8-half (16B) loads.
__device__ __forceinline__ v16h load_b_frag_nmajor(const _Float16* __restrict__ base,
                                                   int stride, int lane) {
  const _Float16* p = base + (size_t)(lane & 15) * stride + ((lane >> 4) << 3);
  v8h lo = *(const v8h*)(p);
  v8h hi = *(const v8h*)(p + 16);
  v16h b;
#pragma unroll
  for (int i = 0; i < 8; ++i) { b[i] = lo[i]; b[i + 8] = hi[i]; }
  return b;
}

// ---------------------------------------------------------------------------
// Kernel 1: fp32 -> f16 conversion, 4 elements/thread.
// ---------------------------------------------------------------------------
__global__ void k_f32_to_f16(const float* __restrict__ in,
                             _Float16* __restrict__ out, long n) {
  long i = ((long)blockIdx.x * blockDim.x + threadIdx.x) * 4;
  if (i + 3 < n) {
    v4f x = *(const v4f*)(in + i);
#pragma unroll
    for (int j = 0; j < 4; ++j) out[i + j] = (_Float16)x[j];
  }
}

// ---------------------------------------------------------------------------
// Shared GEMM core: 128x128 block tile, 8 waves x (16 rows x 128 cols),
// K-step 32, double-buffered LDS fed by async copies.  acc must be zeroed.
// ---------------------------------------------------------------------------
__device__ __forceinline__ void gemm_core(const _Float16* __restrict__ A,
                                          const _Float16* __restrict__ W,
                                          int m0, int n0, int tid,
                                          _Float16 (*As)[128][40],
                                          _Float16 (*Bs)[128][40],
                                          v8f acc[8]) {
  const int lane = tid & 31;
  const int wave = tid >> 5;
  const int ldrow = tid >> 1, ldseg = (tid & 1) << 4;   // 16 halfs/thread

  const _Float16* ga = &A[(size_t)(m0 + ldrow) * EE + ldseg];
  const _Float16* gb = &W[(size_t)(n0 + ldrow) * EE + ldseg];

  // Prime buffer 0 with the k0=0 tiles.
  async_copy16(ga + 0, &As[0][ldrow][ldseg + 0]);
  async_copy16(ga + 8, &As[0][ldrow][ldseg + 8]);
  async_copy16(gb + 0, &Bs[0][ldrow][ldseg + 0]);
  async_copy16(gb + 8, &Bs[0][ldrow][ldseg + 8]);
  wait_async0();
  __syncthreads();

  int cur = 0;
  for (int k0 = 0; k0 < EE; k0 += 32) {
    const int nxt = cur ^ 1;
    if (k0 + 32 < EE) {   // async-fill the other buffer while we compute
      async_copy16(ga + k0 + 32 + 0, &As[nxt][ldrow][ldseg + 0]);
      async_copy16(ga + k0 + 32 + 8, &As[nxt][ldrow][ldseg + 8]);
      async_copy16(gb + k0 + 32 + 0, &Bs[nxt][ldrow][ldseg + 0]);
      async_copy16(gb + k0 + 32 + 8, &Bs[nxt][ldrow][ldseg + 8]);
    }

    v16h a = load_a_frag(&As[cur][wave * 16][0], 40, lane);
    v16h bf[8];
#pragma unroll
    for (int nb = 0; nb < 8; ++nb)
      bf[nb] = load_b_frag_nmajor(&Bs[cur][nb * 16][0], 40, lane);
#pragma unroll
    for (int nb = 0; nb < 8; ++nb)
      acc[nb] = wmma_f16(a, bf[nb], acc[nb]);

    wait_async0();        // our async stores into `nxt` have landed
    __syncthreads();      // everyone done reading `cur`, `nxt` visible
    cur = nxt;
  }
}

// ---------------------------------------------------------------------------
// Kernel 2: QKV GEMM.  A = query_f16 [B*S, E], W = Wqkv_f16 [3E, E].
// Scatters into q [B,H,S,HD], k [B,H,S,HD], vT [B,H,HD,S].
// ---------------------------------------------------------------------------
__global__ __launch_bounds__(256)
void k_gemm_qkv(const _Float16* __restrict__ A,
                const _Float16* __restrict__ W,
                const float* __restrict__ bias,
                _Float16* __restrict__ qh,
                _Float16* __restrict__ kh,
                _Float16* __restrict__ vT) {
  __shared__ _Float16 As[2][128][40];
  __shared__ _Float16 Bs[2][128][40];

  const int tid = threadIdx.x;
  const int lane = tid & 31, wave = tid >> 5;
  const int m0 = blockIdx.x * 128;
  const int n0 = blockIdx.y * 128;

  v8f acc[8];
#pragma unroll
  for (int i = 0; i < 8; ++i)
#pragma unroll
    for (int j = 0; j < 8; ++j) acc[i][j] = 0.f;

  gemm_core(A, W, m0, n0, tid, As, Bs, acc);

  // C/D layout: element vg -> row = vg + (lane>>4)*8, col = lane&15.
#pragma unroll
  for (int nb = 0; nb < 8; ++nb) {
#pragma unroll
    for (int vg = 0; vg < 8; ++vg) {
      int row = m0 + wave * 16 + vg + ((lane >> 4) << 3);
      int col = n0 + nb * 16 + (lane & 15);
      float val = acc[nb][vg] + bias[col];
      int seg = col >> 11;           // col / E
      int cw  = col & (EE - 1);
      int h = cw >> 7, d = cw & (HDD - 1);
      int b = row >> 11, s = row & (SS - 1);
      _Float16 hv = (_Float16)val;
      size_t headbase = ((size_t)b * HH + h);
      if (seg == 0)      qh[(headbase * SS + s) * HDD + d] = hv;
      else if (seg == 1) kh[(headbase * SS + s) * HDD + d] = hv;
      else               vT[(headbase * HDD + d) * SS + s] = hv;
    }
  }
}

// ---------------------------------------------------------------------------
// Kernel 3: flash attention.  Block = (b,h, 128 q-rows); 8 waves x 16 rows.
// ---------------------------------------------------------------------------
__global__ __launch_bounds__(256)
void k_flash_attn(const _Float16* __restrict__ qh,
                  const _Float16* __restrict__ kh,
                  const _Float16* __restrict__ vT,
                  _Float16* __restrict__ ctx) {
  __shared__ _Float16 Ps[8][16][72];   // per-wave P tile (16x64, padded)

  const int tid  = threadIdx.x;
  const int lane = tid & 31;
  const int wave = tid >> 5;
  const int bh = blockIdx.x;
  const int b = bh / HH, h = bh % HH;
  const int q0 = blockIdx.y * 128;
  const float scale = 11.313708498984761f;   // sqrt(128) (reference multiplies)

  const _Float16* Q  = qh + ((size_t)b * HH + h) * SS * HDD;   // [S][HD]
  const _Float16* K  = kh + ((size_t)b * HH + h) * SS * HDD;   // [S][HD]
  const _Float16* Vt = vT + ((size_t)b * HH + h) * HDD * SS;   // [HD][S]

  v16h qa[4];
#pragma unroll
  for (int ks = 0; ks < 4; ++ks)
    qa[ks] = load_a_frag(Q + (size_t)(q0 + wave * 16) * HDD + ks * 32, HDD, lane);

  v8f o[8];
  float m_i[8], l_i[8];
#pragma unroll
  for (int i = 0; i < 8; ++i) {
    m_i[i] = -1e30f; l_i[i] = 0.f;
#pragma unroll
    for (int j = 0; j < 8; ++j) o[i][j] = 0.f;
  }

  for (int kt = 0; kt < SS; kt += 64) {
    if (kt + 64 < SS) {
      __builtin_prefetch(K + (size_t)(kt + 64) * HDD + (lane << 2), 0, 1);
      __builtin_prefetch(Vt + (size_t)(lane) * SS + kt + 64, 0, 1);
    }

    // ---- scores: 16 q-rows x 64 keys; preload all 4 B-frags per K-step ----
    v8f sc[4];
#pragma unroll
    for (int nb = 0; nb < 4; ++nb)
#pragma unroll
      for (int j = 0; j < 8; ++j) sc[nb][j] = 0.f;
#pragma unroll
    for (int ks = 0; ks < 4; ++ks) {
      v16h kb[4];
#pragma unroll
      for (int nb = 0; nb < 4; ++nb)
        kb[nb] = load_b_frag_nmajor(K + (size_t)(kt + nb * 16) * HDD + ks * 32,
                                    HDD, lane);
#pragma unroll
      for (int nb = 0; nb < 4; ++nb)
        sc[nb] = wmma_f16(qa[ks], kb[nb], sc[nb]);
    }

    // ---- online softmax (rows spread over the 16 lanes of each half) ------
#pragma unroll
    for (int vg = 0; vg < 8; ++vg) {
      float mx = -1e30f;
#pragma unroll
      for (int nb = 0; nb < 4; ++nb) mx = fmaxf(mx, sc[nb][vg] * scale);
#pragma unroll
      for (int off = 1; off < 16; off <<= 1) mx = fmaxf(mx, __shfl_xor(mx, off, 32));
      float mn   = fmaxf(m_i[vg], mx);
      float corr = __expf(m_i[vg] - mn);
#pragma unroll
      for (int db = 0; db < 8; ++db) o[db][vg] *= corr;
      l_i[vg] *= corr;
      m_i[vg] = mn;
      float rs = 0.f;
      int prow = vg + ((lane >> 4) << 3);
#pragma unroll
      for (int nb = 0; nb < 4; ++nb) {
        float p = __expf(sc[nb][vg] * scale - mn);
        rs += p;
        Ps[wave][prow][nb * 16 + (lane & 15)] = (_Float16)p;
      }
#pragma unroll
      for (int off = 1; off < 16; off <<= 1) rs += __shfl_xor(rs, off, 32);
      l_i[vg] += rs;
    }
    __syncthreads();   // all waves at same kt; orders Ps stores vs. frag loads

    // ---- O += P(16x64) @ V(64x128); V^T is N-major over d -----------------
#pragma unroll
    for (int ks2 = 0; ks2 < 2; ++ks2) {
      v16h pa = load_a_frag(&Ps[wave][0][ks2 * 32], 72, lane);
      v16h vb[8];
#pragma unroll
      for (int db = 0; db < 8; ++db)
        vb[db] = load_b_frag_nmajor(Vt + (size_t)(db * 16) * SS + kt + ks2 * 32,
                                    SS, lane);
#pragma unroll
      for (int db = 0; db < 8; ++db)
        o[db] = wmma_f16(pa, vb[db], o[db]);
    }
    __syncthreads();
  }

  // ---- normalize + write context [B,S,E] ----------------------------------
#pragma unroll
  for (int db = 0; db < 8; ++db) {
#pragma unroll
    for (int vg = 0; vg < 8; ++vg) {
      int row = q0 + wave * 16 + vg + ((lane >> 4) << 3);
      int d   = db * 16 + (lane & 15);
      float val = o[db][vg] / l_i[vg];
      ctx[((size_t)b * SS + row) * EE + h * HDD + d] = (_Float16)val;
    }
  }
}

// ---------------------------------------------------------------------------
// Kernel 4: output projection.  out(f32) = ctx_f16 @ Wproj^T + bproj.
// ---------------------------------------------------------------------------
__global__ __launch_bounds__(256)
void k_gemm_proj(const _Float16* __restrict__ A,
                 const _Float16* __restrict__ W,
                 const float* __restrict__ bias,
                 float* __restrict__ out) {
  __shared__ _Float16 As[2][128][40];
  __shared__ _Float16 Bs[2][128][40];

  const int tid = threadIdx.x;
  const int lane = tid & 31, wave = tid >> 5;
  const int m0 = blockIdx.x * 128;
  const int n0 = blockIdx.y * 128;

  v8f acc[8];
#pragma unroll
  for (int i = 0; i < 8; ++i)
#pragma unroll
    for (int j = 0; j < 8; ++j) acc[i][j] = 0.f;

  gemm_core(A, W, m0, n0, tid, As, Bs, acc);

#pragma unroll
  for (int nb = 0; nb < 8; ++nb) {
#pragma unroll
    for (int vg = 0; vg < 8; ++vg) {
      int row = m0 + wave * 16 + vg + ((lane >> 4) << 3);
      int col = n0 + nb * 16 + (lane & 15);
      out[(size_t)row * EE + col] = acc[nb][vg] + bias[col];
    }
  }
}

// ---------------------------------------------------------------------------
extern "C" void kernel_launch(void* const* d_in, const int* in_sizes, int n_in,
                              void* d_out, int out_size, void* d_ws, size_t ws_size,
                              hipStream_t stream) {
  (void)in_sizes; (void)n_in; (void)out_size; (void)ws_size;

  const float* query = (const float*)d_in[0];
  // d_in[1] = key, d_in[2] = value: unused by the reference math.
  const float* Wqkv  = (const float*)d_in[3];
  const float* bqkv  = (const float*)d_in[4];
  const float* Wproj = (const float*)d_in[5];
  const float* bproj = (const float*)d_in[6];
  float* out = (float*)d_out;

  const size_t nq     = (size_t)BB * SS * EE;   // 8388608
  const size_t nwqkv  = (size_t)3 * EE * EE;    // 12582912
  const size_t nwproj = (size_t)EE * EE;        // 4194304

  _Float16* qf16    = (_Float16*)d_ws;
  _Float16* wqkv16  = qf16 + nq;
  _Float16* wproj16 = wqkv16 + nwqkv;
  _Float16* qh      = wproj16 + nwproj;   // [B,H,S,HD]
  _Float16* kh      = qh + nq;            // [B,H,S,HD]
  _Float16* vT      = kh + nq;            // [B,H,HD,S]
  _Float16* ctx     = vT + nq;            // [B,S,E]

  {
    long n;
    n = (long)nq;
    k_f32_to_f16<<<(unsigned)((n / 4 + 255) / 256), 256, 0, stream>>>(query, qf16, n);
    n = (long)nwqkv;
    k_f32_to_f16<<<(unsigned)((n / 4 + 255) / 256), 256, 0, stream>>>(Wqkv, wqkv16, n);
    n = (long)nwproj;
    k_f32_to_f16<<<(unsigned)((n / 4 + 255) / 256), 256, 0, stream>>>(Wproj, wproj16, n);
  }

  // 2) QKV projection: [4096 x 2048] @ [2048 x 6144]
  k_gemm_qkv<<<dim3((BB * SS) / 128, (3 * EE) / 128), 256, 0, stream>>>(
      qf16, wqkv16, bqkv, qh, kh, vT);

  // 3) flash attention over 32 (b,h) pairs x 16 q-tiles
  k_flash_attn<<<dim3(BB * HH, SS / 128), 256, 0, stream>>>(qh, kh, vT, ctx);

  // 4) output projection: [4096 x 2048] @ [2048 x 2048] -> fp32 out
  k_gemm_proj<<<dim3((BB * SS) / 128, EE / 128), 256, 0, stream>>>(
      ctx, wproj16, bproj, out);
}
